// CRF_51573967290477
// MI455X (gfx1250) — matrix operand order repaired and checked by reading, not compiled, not run on hardware
//
#include <hip/hip_runtime.h>
#include <hip/hip_bf16.h>

#define T_LEN 16384
#define NT    1024
#define NWG   16          // workgroups (one cluster's worth, one per WGP)
#define COLS  64          // output columns owned per workgroup
#define BLK   128         // 4 waves per workgroup
#define PITCH 1032        // f16 elements per eT row (1024 + 8 pad -> no LDS bank clash)
#define CH    128         // trans rows staged per TDM chunk
#define NCHUNK (NT / CH)

typedef _Float16 h16v __attribute__((ext_vector_type(16)));
typedef _Float16 h8v  __attribute__((ext_vector_type(8)));
typedef float    f8v  __attribute__((ext_vector_type(8)));
typedef unsigned int u32x4 __attribute__((ext_vector_type(4)));
typedef int i32x8 __attribute__((ext_vector_type(8)));
typedef int i32x4 __attribute__((ext_vector_type(4)));

// ---------------------------------------------------------------------------
// TDM: DMA a [rows x COLS] fp32 tile (row stride NT floats) from global into
// LDS (contiguous row-major), per CDNA5 D# layout (ISA 08 §8).  Tracked with
// TENSORcnt.  Issued by one wave; EXEC is ignored by tensor ops.
// This toolchain exposes the 6-arg builtin: (g0, g1, g2, g3, g4, cpol).
// ---------------------------------------------------------------------------
__device__ __forceinline__ void tdm_load_rows(const float* gsrc,
                                              const float* lds_dst, int rows) {
  const unsigned long long ga = (unsigned long long)(size_t)gsrc;
  const unsigned int la = (unsigned int)(size_t)lds_dst;  // addr[31:0] = LDS offset
  u32x4 g0;
  g0[0] = 1u;                                         // count=1, user-mode
  g0[1] = la;                                         // lds_addr
  g0[2] = (unsigned int)(ga & 0xffffffffu);           // global_addr[31:0]
  g0[3] = (unsigned int)((ga >> 32) & 0x01ffffffu)    // global_addr[56:32]
        | (2u << 30);                                 // type=2 ("image")
  i32x8 g1;
  g1[0] = 0x00020000;                                 // data_size=2 (4 bytes)
  g1[1] = (int)((NT & 0xffff) << 16);                 // tensor_dim0[15:0]
  g1[2] = (int)((NT >> 16) & 0xffff)                  // tensor_dim0[31:16]
        | (int)((NT & 0xffff) << 16);                 // tensor_dim1[15:0]
  g1[3] = (int)((NT >> 16) & 0xffff)                  // tensor_dim1[31:16]
        | (COLS << 16);                               // tile_dim0
  g1[4] = rows & 0xffff;                              // tile_dim1 (tile_dim2=0)
  g1[5] = NT;                                         // tensor_dim0_stride[31:0]
  g1[6] = 0;                                          // stride hi / dim1_stride lo
  g1[7] = 0;
  i32x4 z4 = {0, 0, 0, 0};                            // 2D tensor: groups 2/3 unused
  i32x8 z8 = {0, 0, 0, 0, 0, 0, 0, 0};
  __builtin_amdgcn_tensor_load_to_lds(g0, g1, z4, z4, z8, 0);
}

// Cluster barrier (CDNA5) + agent-scope atomic sense barrier fallback.
// s_barrier_signal/wait -3 are NOPs when ClusterID==0, so this is safe for a
// plain 16-block launch and becomes a fast HW barrier under a cluster launch.
__device__ __forceinline__ void grid_barrier(int* cnt, int* gen) {
  __syncthreads();
  if (threadIdx.x == 0) {
    __threadfence();
    __builtin_amdgcn_s_cluster_barrier();
    int g = __hip_atomic_load(gen, __ATOMIC_RELAXED, __HIP_MEMORY_SCOPE_AGENT);
    int arrived = __hip_atomic_fetch_add(cnt, 1, __ATOMIC_ACQ_REL,
                                         __HIP_MEMORY_SCOPE_AGENT);
    if (arrived == NWG - 1) {
      __hip_atomic_store(cnt, 0, __ATOMIC_RELAXED, __HIP_MEMORY_SCOPE_AGENT);
      __hip_atomic_fetch_add(gen, 1, __ATOMIC_RELEASE, __HIP_MEMORY_SCOPE_AGENT);
    } else {
      while (__hip_atomic_load(gen, __ATOMIC_ACQUIRE,
                               __HIP_MEMORY_SCOPE_AGENT) == g)
        __builtin_amdgcn_s_sleep(1);
    }
    __threadfence();
  }
  __syncthreads();
}

// Gold-path score + barrier state init (runs first on the stream).
__global__ void crf_score_init_kernel(const float* __restrict__ emit,
                                      const int*   __restrict__ y,
                                      const float* __restrict__ trans,
                                      const float* __restrict__ strans,
                                      const float* __restrict__ etrans,
                                      float* __restrict__ score_out,
                                      int* cnt, int* gen) {
  __shared__ float red[256];
  const int tid = threadIdx.x;
  if (tid == 0) { *cnt = 0; *gen = 0; }
  float s = 0.f;
  for (int t = tid; t < T_LEN; t += 256) {
    const int yt = y[t];
    s += emit[(size_t)t * NT + yt];
    if (t >= 1) s += trans[(size_t)y[t - 1] * NT + yt];
  }
  red[tid] = s; __syncthreads();
  for (int off = 128; off > 0; off >>= 1) {
    if (tid < off) red[tid] += red[tid + off];
    __syncthreads();
  }
  if (tid == 0)
    *score_out = red[0] + strans[y[0]] + etrans[y[T_LEN - 1]];
}

// Persistent scan: 16 WGs, each holds a 64-column f16 slice of exp(trans)
// in LDS (filled via double-buffered TDM tile loads) and advances alpha via
// WMMA GEMV + log/exp, syncing every step.
__global__ void __launch_bounds__(BLK)
crf_scan_kernel(const float* __restrict__ emit,
                const float* __restrict__ trans,
                const float* __restrict__ strans,
                const float* __restrict__ etrans,
                const float* __restrict__ score,
                float* __restrict__ galpha,      // [2][NT] exchange buffer
                int* cnt, int* gen,
                float* __restrict__ out) {
  extern __shared__ char smem[];
  _Float16* eT  = (_Float16*)smem;               // [COLS][PITCH]  ~132 KB
  _Float16* p16 = eT + COLS * PITCH;             // [NT]            2 KB
  float* alpha  = (float*)(p16 + NT);            // [NT]            4 KB
  float* red    = alpha + NT;                    // [BLK]           0.5 KB
  float* stage  = red + BLK;                     // [2][CH*COLS]    64 KB

  const int tid  = threadIdx.x;
  const int wg   = blockIdx.x;
  const int base = wg * COLS;
  const int lane = tid & 31;
  const int wave = tid >> 5;
  const int nloc = lane & 15;
  const int lrow = wave * 16 + nloc;             // local column 0..63
  const int gcol = base + lrow;                  // global column
  const int hi   = (lane >= 16) ? 1 : 0;

  // ---- Fill eT[r][j] = exp(trans[j][base+r]) as f16, TDM-pipelined. -------
  if (tid < 32) tdm_load_rows(trans + base, stage, CH);   // chunk 0
  for (int c = 0; c < NCHUNK; ++c) {
    if (tid < 32) {
      if (c + 1 < NCHUNK) {
        tdm_load_rows(trans + (size_t)(c + 1) * CH * NT + base,
                      stage + ((c + 1) & 1) * (CH * COLS), CH);
        __builtin_amdgcn_s_wait_tensorcnt(1);    // chunk c done, c+1 in flight
      } else {
        __builtin_amdgcn_s_wait_tensorcnt(0);    // last chunk done
      }
    }
    __syncthreads();
    const float* sb = stage + (c & 1) * (CH * COLS);
    for (int idx = tid; idx < CH * COLS; idx += BLK) {
      const int r = idx & (COLS - 1);            // column within slice
      const int j = idx >> 6;                    // row within chunk
      eT[r * PITCH + (c * CH + j)] = (_Float16)__expf(sb[j * COLS + r]);
    }
    __syncthreads();
  }

  // alpha0 = strans + emit[0]
  for (int j = tid; j < NT; j += BLK)
    alpha[j] = strans[j] + emit[j];
  __syncthreads();

  for (int t = 1; t < T_LEN; ++t) {
    // m = max(alpha): wave32 shfl butterfly + one cross-wave LDS stage.
    float lm = -3.4e38f;
    for (int j = tid; j < NT; j += BLK) lm = fmaxf(lm, alpha[j]);
#pragma unroll
    for (int off = 16; off > 0; off >>= 1)
      lm = fmaxf(lm, __shfl_xor(lm, off, 32));
    if (lane == 0) red[wave] = lm;
    __syncthreads();
    const float m = fmaxf(fmaxf(red[0], red[1]), fmaxf(red[2], red[3]));

    // p = exp(alpha - m) in f16
    for (int j = tid; j < NT; j += BLK)
      p16[j] = (_Float16)__expf(alpha[j] - m);
    __syncthreads();

    // GEMV slice via v_wmma_f32_16x16x32_f16: this wave computes 16 columns.
    // A (16x32, rows replicated with p-chunk): lanes0-15 K{0-7,16-23},
    // lanes16-31 K{8-15,24-31}.  B (32x16): lane n holds col n%16, same K
    // split as A: K{0-7,16-23} (lanes<16) / K{8-15,24-31} (lanes>=16).
    // Two accumulators so consecutive WMMAs are independent.
    f8v acc0 = {0.f, 0.f, 0.f, 0.f, 0.f, 0.f, 0.f, 0.f};
    f8v acc1 = {0.f, 0.f, 0.f, 0.f, 0.f, 0.f, 0.f, 0.f};
    const _Float16* brow = eT + (size_t)lrow * PITCH + (hi ? 8 : 0);
#pragma unroll 2
    for (int j0 = 0; j0 < NT; j0 += 64) {
      const _Float16* pa0 = p16 + j0 + (hi ? 8 : 0);
      h8v a00 = *(const h8v*)(pa0);
      h8v a01 = *(const h8v*)(pa0 + 16);
      h8v b00 = *(const h8v*)(brow + j0);
      h8v b01 = *(const h8v*)(brow + j0 + 16);
      const _Float16* pa1 = pa0 + 32;
      h8v a10 = *(const h8v*)(pa1);
      h8v a11 = *(const h8v*)(pa1 + 16);
      h8v b10 = *(const h8v*)(brow + j0 + 32);
      h8v b11 = *(const h8v*)(brow + j0 + 48);
      h16v A0, B0, A1, B1;
#pragma unroll
      for (int i = 0; i < 8; ++i) {
        A0[i] = a00[i]; A0[i + 8] = a01[i];
        B0[i] = b00[i]; B0[i + 8] = b01[i];
        A1[i] = a10[i]; A1[i + 8] = a11[i];
        B1[i] = b10[i]; B1[i + 8] = b11[i];
      }
      acc0 = __builtin_amdgcn_wmma_f32_16x16x32_f16(
          false, A0, false, B0, (short)0, acc0, false, false);
      acc1 = __builtin_amdgcn_wmma_f32_16x16x32_f16(
          false, A1, false, B1, (short)0, acc1, false, false);
    }
    const float sdot = acc0[0] + acc1[0];  // C rows identical (replicated A)

    const float e = emit[(size_t)t * NT + gcol];
    if (t + 1 < T_LEN)
      __builtin_prefetch(&emit[(size_t)(t + 1) * NT + gcol], 0, 1);
    const float anew = e + m + __logf(sdot);
    const int buf = (t & 1) * NT;                // double-buffered exchange
    if (lane < 16) {
      alpha[gcol] = anew;
      galpha[buf + gcol] = anew;
    }
    grid_barrier(cnt, gen);
    for (int j = tid; j < NT; j += BLK)
      if ((j >> 6) != wg) alpha[j] = galpha[buf + j];
    __syncthreads();
  }

  // logZ = logsumexp(alpha + etrans); out = logZ - score  (WG0 only)
  if (wg == 0) {
    float lm = -3.4e38f;
    for (int j = tid; j < NT; j += BLK) lm = fmaxf(lm, alpha[j] + etrans[j]);
#pragma unroll
    for (int off = 16; off > 0; off >>= 1)
      lm = fmaxf(lm, __shfl_xor(lm, off, 32));
    if (lane == 0) red[wave] = lm;
    __syncthreads();
    const float m2 = fmaxf(fmaxf(red[0], red[1]), fmaxf(red[2], red[3]));
    __syncthreads();
    float ls = 0.f;
    for (int j = tid; j < NT; j += BLK)
      ls += __expf(alpha[j] + etrans[j] - m2);
#pragma unroll
    for (int off = 16; off > 0; off >>= 1)
      ls += __shfl_xor(ls, off, 32);
    if (lane == 0) red[wave] = ls;
    __syncthreads();
    if (tid == 0)
      out[0] = m2 + __logf(red[0] + red[1] + red[2] + red[3]) - score[0];
  }
}

extern "C" void kernel_launch(void* const* d_in, const int* in_sizes, int n_in,
                              void* d_out, int out_size, void* d_ws, size_t ws_size,
                              hipStream_t stream) {
  const float* emit   = (const float*)d_in[0];
  const int*   y      = (const int*)d_in[1];
  const float* trans  = (const float*)d_in[2];
  const float* strans = (const float*)d_in[3];
  const float* etrans = (const float*)d_in[4];
  float* out = (float*)d_out;

  char* ws = (char*)d_ws;
  float* score  = (float*)(ws + 0);
  int*   cnt    = (int*)(ws + 64);
  int*   gen    = (int*)(ws + 128);
  float* galpha = (float*)(ws + 192);   // 2*NT floats

  crf_score_init_kernel<<<1, 256, 0, stream>>>(emit, y, trans, strans, etrans,
                                               score, cnt, gen);
  const size_t lds = (size_t)(COLS * PITCH + NT) * sizeof(_Float16)
                   + (size_t)(NT + BLK + 2 * CH * COLS) * sizeof(float);
  crf_scan_kernel<<<NWG, BLK, lds, stream>>>(emit, trans, strans, etrans,
                                             score, galpha, cnt, gen, out);
}
